// GTU_layer_14276471292462
// MI455X (gfx1250) — matrix operand order repaired and checked by reading, not compiled, not run on hardware
//
#include <hip/hip_runtime.h>

// ---------------------------------------------------------------------------
// GTU / TNN layer for MI455X (gfx1250, wave32, WMMA, async global->LDS)
// B=4, N=4096, E=256, H=8, D1=768, HD=96, RPE=32, LAYERS=3
// ---------------------------------------------------------------------------

#define B_      4
#define N_      4096
#define E_      256
#define H_      8
#define D1_     768
#define HD_     96
#define RPE_    32
#define LAYERS_ 3
#define N2_     8192          // 2n (power of two -> mod is AND)
#define EPS_    1e-8f

typedef __attribute__((ext_vector_type(16))) __bf16 v16bf;
typedef __attribute__((ext_vector_type(8)))  float  v8f;

union Frag {
  v16bf v;
  unsigned short u[16];
  uint4 q[2];
};

static __device__ __forceinline__ unsigned short f2bf(float f) {
  unsigned int x = __float_as_uint(f);
  x += 0x7FFFu + ((x >> 16) & 1u);          // round-to-nearest-even
  return (unsigned short)(x >> 16);
}
static __device__ __forceinline__ float silu_f(float x) {
  return x / (1.0f + __expf(-x));
}

#define WMMA_BF16(Afr, Bfr, C) \
  __builtin_amdgcn_wmma_f32_16x16x32_bf16(false, (Afr), false, (Bfr), (short)0, (C), false, false)

// Async global -> LDS copy, 16 bytes per lane (GV addressing, ASYNCcnt).
static __device__ __forceinline__ void async_copy_b128(unsigned lds_off,
                                                       const void* gaddr) {
  asm volatile("global_load_async_to_lds_b128 %0, %1, off"
               :: "v"(lds_off), "v"(gaddr) : "memory");
}
static __device__ __forceinline__ void wait_async0() {
  asm volatile("s_wait_asynccnt 0x0" ::: "memory");
}
static __device__ __forceinline__ unsigned lds_addr_of(const void* p) {
  // Flat aperture: LDS offset is the low 32 bits of the generic address
  return (unsigned)(unsigned long long)p;
}

// A fragment (16x32 bf16, row-major source, ld in elements).
// lane<16: K 0..15 ; lane>=16: K 16..31 (contig) ; row = lane&15
static __device__ __forceinline__ void load_fragA_rm(
    const unsigned short* __restrict__ base, int ld, int lane, Frag& f) {
  const int row = lane & 15;
  const int kb  = (lane >> 4) * 16;
  const unsigned short* r = base + (size_t)row * ld + kb;
  f.q[0] = *(const uint4*)(r);
  f.q[1] = *(const uint4*)(r + 8);
}
// B fragment (32x16 bf16) from weight matrix W[Nc][K] row-major:
// B[k][n] = W[j0+n][k]. lane: n = lane&15, K range = (lane>>4)*16 .. +15 (contig)
static __device__ __forceinline__ void load_fragB_rm(
    const unsigned short* __restrict__ Wrow, int ld, int lane, Frag& f) {
  const int col = lane & 15;
  const int kb  = (lane >> 4) * 16;
  const uint4* p = (const uint4*)(Wrow + (size_t)col * ld + kb);
  f.q[0] = p[0];
  f.q[1] = p[1];
}

// ---------------------------------------------------------------------------
// float -> bf16 conversion
// ---------------------------------------------------------------------------
__global__ void k_cvt_bf16(const float* __restrict__ src,
                           unsigned short* __restrict__ dst, int n) {
  int i = blockIdx.x * blockDim.x + threadIdx.x;
  const int stride = gridDim.x * blockDim.x;
  for (; i < n; i += stride) dst[i] = f2bf(src[i]);
}

// ---------------------------------------------------------------------------
// RPE MLP: positions [0, 1..n-1, 0, -(n-1)..-1] -> a[j][p]  (j = h*HD+d)
// ---------------------------------------------------------------------------
__global__ void k_rpe(const float* __restrict__ pos_w, const float* __restrict__ pos_b,
                      const float* __restrict__ lw,    const float* __restrict__ lb,
                      const float* __restrict__ ow,    const float* __restrict__ ob,
                      unsigned short* __restrict__ a_bf) {
  const int p = blockIdx.x * blockDim.x + threadIdx.x;
  if (p >= N2_) return;
  float iv;
  if (p == 0 || p == N_)  iv = 0.0f;
  else if (p < N_)        iv = (float)p;
  else                    iv = -(float)(N2_ - p);

  const float inv_sqrt_d = 0.17677669529663687f;  // 32^-0.5
  float h[RPE_], t[RPE_];
  #pragma unroll
  for (int r = 0; r < RPE_; ++r) h[r] = iv * pos_w[r] + pos_b[r];

  for (int L = 0; L < LAYERS_; ++L) {
    float ss = 0.f;
    #pragma unroll
    for (int r = 0; r < RPE_; ++r) ss += h[r] * h[r];
    const float inv = 1.0f / (__fsqrt_rn(ss) * inv_sqrt_d + EPS_);
    #pragma unroll
    for (int r = 0; r < RPE_; ++r) { float y = h[r] * inv; t[r] = y > 0.f ? y : 0.f; }
    const float* W  = lw + L * RPE_ * RPE_;
    const float* bb = lb + L * RPE_;
    #pragma unroll
    for (int r = 0; r < RPE_; ++r) {
      float acc = bb[r];
      #pragma unroll
      for (int c = 0; c < RPE_; ++c) acc += W[r * RPE_ + c] * t[c];
      h[r] = acc;
    }
  }
  {
    float ss = 0.f;
    #pragma unroll
    for (int r = 0; r < RPE_; ++r) ss += h[r] * h[r];
    const float inv = 1.0f / (__fsqrt_rn(ss) * inv_sqrt_d + EPS_);
    #pragma unroll
    for (int r = 0; r < RPE_; ++r) { float y = h[r] * inv; t[r] = y > 0.f ? y : 0.f; }
  }
  for (int j = 0; j < D1_; ++j) {
    float acc = ob[j];
    const float* W = ow + j * RPE_;
    #pragma unroll
    for (int c = 0; c < RPE_; ++c) acc += W[c] * t[c];
    a_bf[(size_t)j * N2_ + p] = f2bf(acc);
  }
}

// ---------------------------------------------------------------------------
// Fused u/v GEMM: u = silu(x@u_w^T+u_b) (f32, [m][j]);
//                 v = silu(x@v_w^T+v_b) (bf16, scattered to [b*D1+j][n])
// One wave per 16x16 tile; A fragments shared by both weight matrices.
// ---------------------------------------------------------------------------
__global__ __launch_bounds__(256)
void k_gemm_uv(const unsigned short* __restrict__ xbf,
               const unsigned short* __restrict__ uW,
               const unsigned short* __restrict__ vW,
               const float* __restrict__ u_b, const float* __restrict__ v_b,
               float* __restrict__ u_out, unsigned short* __restrict__ v_bf) {
  const int lane = threadIdx.x & 31;
  const int wave = threadIdx.x >> 5;
  const int tile = blockIdx.x * 8 + wave;
  const int JT = D1_ / 16;                       // 48
  const int m0 = (tile / JT) * 16;
  const int j0 = (tile % JT) * 16;
  v8f accU = {}; v8f accV = {};
  for (int k0 = 0; k0 < E_; k0 += 32) {
    Frag a, bu, bv;
    load_fragA_rm(xbf + (size_t)m0 * E_ + k0, E_, lane, a);
    load_fragB_rm(uW  + (size_t)j0 * E_ + k0, E_, lane, bu);
    load_fragB_rm(vW  + (size_t)j0 * E_ + k0, E_, lane, bv);
    if (k0 + 32 < E_)
      __builtin_prefetch(xbf + (size_t)m0 * E_ + k0 + 32, 0, 3);
    accU = WMMA_BF16(a.v, bu.v, accU);
    accV = WMMA_BF16(a.v, bv.v, accV);
  }
  const int j    = j0 + (lane & 15);
  const int mrow = m0 + ((lane >> 4) ? 8 : 0);
  const float ub = u_b[j], vb = v_b[j];
  #pragma unroll
  for (int r = 0; r < 8; ++r) {
    const int m = mrow + r;
    u_out[(size_t)m * D1_ + j] = silu_f(accU[r] + ub);
    const int b = m >> 12, n = m & (N_ - 1);
    v_bf[(size_t)(b * D1_ + j) * N_ + n] = f2bf(silu_f(accV[r] + vb));
  }
}

// ---------------------------------------------------------------------------
// Toeplitz mixing via WMMA: out[b][hd][t] = sum_s a[hd][(t-s) & 8191] * v[b][hd][s]
// One workgroup per (h,d); a + 4 batches of v staged into 48 KB LDS via
// CDNA5 async global->LDS copies (ASYNCcnt path, no VGPR round-trip).
// M = t-tile (16), N = batch (4 real lanes of 16), K = s (steps of 32).
// ---------------------------------------------------------------------------
__global__ __launch_bounds__(256)
void k_toeplitz(const unsigned short* __restrict__ a_bf,
                const unsigned short* __restrict__ v_bf,
                float* __restrict__ outc) {
  __shared__ unsigned short a_loc[N2_];          // 16 KB
  __shared__ unsigned short v_loc[B_][N_];       // 32 KB
  const int hd  = blockIdx.x;
  const int tid = threadIdx.x;

  // a_loc: 16384 B = 1024 x 16B chunks, 4 per thread
  {
    const unsigned a_base = lds_addr_of(&a_loc[0]);
    const char* g = (const char*)(a_bf + (size_t)hd * N2_);
    for (int c = tid; c < (N2_ * 2) / 16; c += 256)
      async_copy_b128(a_base + c * 16, g + c * 16);
  }
  // v_loc: 32768 B = 2048 x 16B chunks, 8 per thread; b-row stride 8192 B
  {
    const unsigned v_base = lds_addr_of(&v_loc[0][0]);
    for (int c = tid; c < 2048; c += 256) {
      const int b = c >> 9, w = c & 511;
      const char* g = (const char*)(v_bf + (size_t)(b * D1_ + hd) * N_);
      async_copy_b128(v_base + (b << 13) + (w << 4), g + (w << 4));
    }
  }
  wait_async0();
  __syncthreads();

  const int lane = tid & 31;
  const int wave = tid >> 5;
  const int tr   = lane & 15;
  const int kb8  = (lane >> 4) * 8;
  const int nb   = lane & 15;
  const int kb16 = (lane >> 4) * 16;

  for (int tt = wave; tt < N_ / 16; tt += 8) {
    const int t0 = tt * 16;
    v8f acc = {};
    for (int s0 = 0; s0 < N_; s0 += 32) {
      Frag a, bfr;
      const int base = t0 + tr - s0;
      #pragma unroll
      for (int i = 0; i < 8; ++i) {
        a.u[i]     = a_loc[(base - (kb16 + i))      & (N2_ - 1)];
        a.u[8 + i] = a_loc[(base - (kb16 + 8 + i)) & (N2_ - 1)];
      }
      if (nb < B_) {
        const uint4* p = (const uint4*)&v_loc[nb][s0 + kb16];
        bfr.q[0] = p[0]; bfr.q[1] = p[1];
      } else {
        bfr.q[0] = make_uint4(0u, 0u, 0u, 0u);
        bfr.q[1] = make_uint4(0u, 0u, 0u, 0u);
      }
      acc = WMMA_BF16(a.v, bfr.v, acc);
    }
    if (nb < B_) {
      float* o = outc + (size_t)(nb * D1_ + hd) * N_ + t0 + ((lane >> 4) ? 8 : 0);
      #pragma unroll
      for (int r = 0; r < 8; ++r) o[r] = acc[r];
    }
  }
}

// ---------------------------------------------------------------------------
// Gate: g[m][j] = bf16( u[m][j] * outc[b*D1+j][n] )
// ---------------------------------------------------------------------------
__global__ void k_gate(const float* __restrict__ u, const float* __restrict__ outc,
                       unsigned short* __restrict__ g_bf) {
  const size_t total = (size_t)B_ * N_ * D1_;
  size_t i = (size_t)blockIdx.x * blockDim.x + threadIdx.x;
  const size_t stride = (size_t)gridDim.x * blockDim.x;
  for (; i < total; i += stride) {
    const size_t m = i / D1_;
    const int j = (int)(i % D1_);
    const int b = (int)(m >> 12);
    const int n = (int)(m & (N_ - 1));
    g_bf[i] = f2bf(u[i] * outc[(size_t)(b * D1_ + j) * N_ + n]);
  }
}

// ---------------------------------------------------------------------------
// o-projection GEMM with fused  z = x - (g@o_w^T + o_b)  epilogue (bf16 out)
// ---------------------------------------------------------------------------
__global__ __launch_bounds__(256)
void k_gemm_o(const unsigned short* __restrict__ gbf,
              const unsigned short* __restrict__ oW,
              const float* __restrict__ o_b, const float* __restrict__ x,
              unsigned short* __restrict__ zbf) {
  const int lane = threadIdx.x & 31;
  const int wave = threadIdx.x >> 5;
  const int tile = blockIdx.x * 8 + wave;
  const int JT = E_ / 16;                        // 16
  const int m0 = (tile / JT) * 16;
  const int j0 = (tile % JT) * 16;
  v8f acc = {};
  for (int k0 = 0; k0 < D1_; k0 += 32) {
    Frag a, b;
    load_fragA_rm(gbf + (size_t)m0 * D1_ + k0, D1_, lane, a);
    load_fragB_rm(oW  + (size_t)j0 * D1_ + k0, D1_, lane, b);
    if (k0 + 32 < D1_)
      __builtin_prefetch(gbf + (size_t)m0 * D1_ + k0 + 32, 0, 3);
    acc = WMMA_BF16(a.v, b.v, acc);
  }
  const int j    = j0 + (lane & 15);
  const int mrow = m0 + ((lane >> 4) ? 8 : 0);
  const float bj = o_b[j];
  #pragma unroll
  for (int r = 0; r < 8; ++r) {
    const int m = mrow + r;
    const float z = x[(size_t)m * E_ + j] - (acc[r] + bj);
    zbf[(size_t)m * E_ + j] = f2bf(z);
  }
}

// ---------------------------------------------------------------------------
// t-projection GEMM: w3 = z@t_w^T + t_b  (f32 out)
// ---------------------------------------------------------------------------
__global__ __launch_bounds__(256)
void k_gemm_t(const unsigned short* __restrict__ zbf,
              const unsigned short* __restrict__ tW,
              const float* __restrict__ t_b, float* __restrict__ w3) {
  const int lane = threadIdx.x & 31;
  const int wave = threadIdx.x >> 5;
  const int tile = blockIdx.x * 8 + wave;
  const int JT = E_ / 16;
  const int m0 = (tile / JT) * 16;
  const int j0 = (tile % JT) * 16;
  v8f acc = {};
  for (int k0 = 0; k0 < E_; k0 += 32) {
    Frag a, b;
    load_fragA_rm(zbf + (size_t)m0 * E_ + k0, E_, lane, a);
    load_fragB_rm(tW  + (size_t)j0 * E_ + k0, E_, lane, b);
    acc = WMMA_BF16(a.v, b.v, acc);
  }
  const int j    = j0 + (lane & 15);
  const int mrow = m0 + ((lane >> 4) ? 8 : 0);
  const float bj = t_b[j];
  #pragma unroll
  for (int r = 0; r < 8; ++r)
    w3[(size_t)(mrow + r) * E_ + j] = acc[r] + bj;
}

// ---------------------------------------------------------------------------
// Final: out = x + silu( w3 / (||w3_row|| * 768^-0.5 + eps) )
// ---------------------------------------------------------------------------
__global__ __launch_bounds__(256)
void k_final(const float* __restrict__ w3, const float* __restrict__ x,
             float* __restrict__ out) {
  __shared__ float red[256];
  const int m = blockIdx.x;
  const int e = threadIdx.x;
  const float val = w3[(size_t)m * E_ + e];
  red[e] = val * val;
  __syncthreads();
  for (int s = 128; s > 0; s >>= 1) {
    if (e < s) red[e] += red[e + s];
    __syncthreads();
  }
  const float rms = __fsqrt_rn(red[0]) * 0.036084391824351615f;  // 768^-0.5
  const float y = val / (rms + EPS_);
  out[(size_t)m * E_ + e] = x[(size_t)m * E_ + e] + silu_f(y);
}

// ---------------------------------------------------------------------------
extern "C" void kernel_launch(void* const* d_in, const int* in_sizes, int n_in,
                              void* d_out, int out_size, void* d_ws, size_t ws_size,
                              hipStream_t stream) {
  (void)in_sizes; (void)n_in; (void)out_size; (void)ws_size;
  const float* x     = (const float*)d_in[0];
  const float* u_w   = (const float*)d_in[1];
  const float* u_b   = (const float*)d_in[2];
  const float* v_w   = (const float*)d_in[3];
  const float* v_b   = (const float*)d_in[4];
  const float* o_w   = (const float*)d_in[5];
  const float* o_b   = (const float*)d_in[6];
  const float* t_w   = (const float*)d_in[7];
  const float* t_b   = (const float*)d_in[8];
  const float* pos_w = (const float*)d_in[9];
  const float* pos_b = (const float*)d_in[10];
  const float* lw    = (const float*)d_in[11];
  const float* lb    = (const float*)d_in[12];
  const float* ow    = (const float*)d_in[13];
  const float* ob    = (const float*)d_in[14];
  float* out = (float*)d_out;

  char* ws = (char*)d_ws;
  size_t off = 0;
  auto take = [&](size_t bytes) -> char* {
    char* p = ws + off;
    off = (off + bytes + 255) & ~(size_t)255;
    return p;
  };
  const size_t MN = (size_t)B_ * N_;                                   // 16384
  unsigned short* xbf  = (unsigned short*)take(MN * E_ * 2);           // 8.4 MB
  unsigned short* uW   = (unsigned short*)take((size_t)D1_ * E_ * 2);
  unsigned short* vW   = (unsigned short*)take((size_t)D1_ * E_ * 2);
  unsigned short* oW   = (unsigned short*)take((size_t)E_ * D1_ * 2);
  unsigned short* tW   = (unsigned short*)take((size_t)E_ * E_ * 2);
  float*          uF   = (float*)take(MN * D1_ * 4);                   // 50 MB
  unsigned short* vbf  = (unsigned short*)take(MN * D1_ * 2);          // 25 MB
  unsigned short* abf  = (unsigned short*)take((size_t)D1_ * N2_ * 2); // 12.6 MB
  float*          outc = (float*)take(MN * D1_ * 4);                   // 50 MB
  unsigned short* gbf  = vbf;   // reuse: v dead after toeplitz
  unsigned short* zbf  = xbf;   // reuse: x_bf dead after gemm_uv
  float*          w3   = outc;  // reuse: outc dead after gate

  k_cvt_bf16<<<4096, 256, 0, stream>>>(x,   xbf, (int)(MN * E_));
  k_cvt_bf16<<<768,  256, 0, stream>>>(u_w, uW, D1_ * E_);
  k_cvt_bf16<<<768,  256, 0, stream>>>(v_w, vW, D1_ * E_);
  k_cvt_bf16<<<768,  256, 0, stream>>>(o_w, oW, E_ * D1_);
  k_cvt_bf16<<<256,  256, 0, stream>>>(t_w, tW, E_ * E_);

  k_rpe<<<N2_ / 256, 256, 0, stream>>>(pos_w, pos_b, lw, lb, ow, ob, abf);

  k_gemm_uv<<<(int)((MN / 16) * (D1_ / 16) / 8), 256, 0, stream>>>(
      xbf, uW, vW, u_b, v_b, uF, vbf);

  k_toeplitz<<<D1_, 256, 0, stream>>>(abf, vbf, outc);

  k_gate<<<8192, 256, 0, stream>>>(uF, outc, gbf);

  k_gemm_o<<<(int)((MN / 16) * (E_ / 16) / 8), 256, 0, stream>>>(
      gbf, oW, o_b, x, zbf);

  k_gemm_t<<<(int)((MN / 16) * (E_ / 16) / 8), 256, 0, stream>>>(
      zbf, tW, t_b, w3);

  k_final<<<(int)MN, 256, 0, stream>>>(w3, x, out);
}
